// TextLoss_42262478192859
// MI455X (gfx1250) — matrix operand order repaired and checked by reading, not compiled, not run on hardware
//
#include <hip/hip_runtime.h>
#include <hip/hip_bf16.h>
#include <math.h>

typedef __attribute__((ext_vector_type(16))) _Float16 v16h;
typedef __attribute__((ext_vector_type(2)))  _Float16 v2h;
typedef __attribute__((ext_vector_type(8)))  float    v8f;

#define HW      (512 * 512)     // per-image pixels
#define NIMG    8
#define NPOLY   1024
#define PN2     256             // PNUM*2 floats per polygon

// ---------------------------------------------------------------------------
// Kernel 1: streaming field losses (cls / dis / norm / angle), memory-bound.
// 1024 blocks x 256 threads; each block owns a 2048-element chunk inside ONE
// image (128 blocks/image) so per-image accumulators reduce deterministically.
// partial[blk*8+q]: q=0 cls, 1 norm, 2 angle, 3 msum, 4 npos, 5 pos_sum, 6 neg_sum
// ---------------------------------------------------------------------------
__global__ __launch_bounds__(256) void field_losses_kernel(
    const float* __restrict__ fy,      // (B,4,H,W)
    const int*   __restrict__ train_m, // (B,H,W)
    const int*   __restrict__ tr_m,    // (B,H,W)
    const float* __restrict__ dfld,    // (B,H,W)
    const float* __restrict__ dirf,    // (B,2,H,W)
    const float* __restrict__ wmat,    // (B,H,W)
    float* __restrict__ partial)
{
    const int blk  = blockIdx.x;       // 0..1023
    const int b    = blk >> 7;         // image index (128 blocks / image)
    const int base = blk * 2048;
    const size_t fyb  = (size_t)b * 4 * HW;
    const size_t dirb = (size_t)b * 2 * HW;

    float cls = 0.f, nrm = 0.f, ang = 0.f, msum = 0.f;
    float npos = 0.f, psum = 0.f, nsum = 0.f;

    #pragma unroll
    for (int it = 0; it < 8; ++it) {
        const int idx = base + it * 256 + threadIdx.x;   // flat over B*HW
        const int hw  = idx - b * HW;

        const float f0 = fy[fyb + hw];
        const float f1 = fy[fyb + HW + hw];
        const float f2 = fy[fyb + 2 * HW + hw];
        const float f3 = fy[fyb + 3 * HW + hw];
        const float tm = (float)train_m[idx];
        const int   tr = tr_m[idx];
        const float d  = dfld[idx];
        const float dx = dirf[dirb + hw];
        const float dy = dirf[dirb + HW + hw];
        const float w  = wmat[idx];

        // --- cls: BCE(fy0, tr>0) * tm
        const float conf = (tr > 0) ? 1.f : 0.f;
        float p = fminf(fmaxf(f0, 1e-7f), 1.f - 1e-7f);
        cls += -(conf * logf(p) + (1.f - conf) * log1pf(-p)) * tm;

        // --- dis: (fy1 - df)^2 * tm, split by label >= 0.001
        float dl = (f1 - d); dl = dl * dl * tm;
        if (d >= 0.001f) { npos += 1.f; psum += dl; } else { nsum += dl; }

        // --- norm: wm * 0.5*((f2-gnx)^2+(f3-gny)^2) * tm
        const float gn   = sqrtf(dx * dx + dy * dy);
        const float ginv = 0.999999f / (gn + 0.001f);
        const float gx = dx * ginv, gy = dy * ginv;
        const float e2 = f2 - gx, e3 = f3 - gy;
        nrm += w * 0.5f * (e2 * e2 + e3 * e3) * tm;

        // --- angle
        const float pn   = sqrtf(f2 * f2 + f3 * f3);
        const float pinv = 0.999999f / (pn + 0.001f);
        const float px = f2 * pinv, py = f3 * pinv;
        const float npn = sqrtf(px * px + py * py);
        const float ngn = sqrtf(gx * gx + gy * gy);
        const float cosv = (px * gx + py * gy) / fmaxf(npn * ngn, 1e-8f);
        const float mm = (tm > 0.f && tr > 0) ? 1.f : 0.f;
        ang  += (1.f - cosv) * mm;
        msum += mm;
    }

    __shared__ float sred[256];
    float vals[7] = {cls, nrm, ang, msum, npos, psum, nsum};
    for (int q = 0; q < 7; ++q) {
        sred[threadIdx.x] = vals[q];
        __syncthreads();
        for (int off = 128; off > 0; off >>= 1) {
            if (threadIdx.x < off) sred[threadIdx.x] += sred[threadIdx.x + off];
            __syncthreads();
        }
        if (threadIdx.x == 0) partial[blk * 8 + q] = sred[0];
        __syncthreads();
    }
}

// ---------------------------------------------------------------------------
// Kernel 2: poly match via WMMA. pred,gt in [0,1) => |d|<1 always, so
// smooth_l1(d) == 0.5 d^2 exactly and
//   128*D_t[s] = 0.5*(Sp_t + Sg) - C_t[s],  C_t[s] = sum_k p_t[k]*g[(k+2s)&255]
// C is a (3 pad 16)x256 x 256x128 matmul per polygon: v_wmma_f32_16x16x32_f16.
// One wave per polygon; kk-outer / st-inner so each A fragment is loaded once
// and 8 independent WMMAs (distinct accumulators) issue back-to-back.
// B fragments: gt duplicated into a 512-half LDS ring packed as dwords, so each
// fragment is 8 contiguous ds_load_b32 (no per-element u16 gather, no mod).
// ---------------------------------------------------------------------------
__global__ __launch_bounds__(32) void poly_match_wmma_kernel(
    const float* __restrict__ p0, const float* __restrict__ p1,
    const float* __restrict__ p2, const float* __restrict__ gt,
    float* __restrict__ minD)
{
    __shared__ _Float16     Ah[16 * PN2];  // rows 0..2 = preds; rows 3..15 unused
    __shared__ unsigned int Bw[256];       // gt halves duplicated: Bw[i] = (g[2i&255], g[(2i+1)&255])

    const int n    = blockIdx.x;
    const int lane = threadIdx.x;
    const float* g  = gt + (size_t)n * PN2;
    const float* q0 = p0 + (size_t)n * PN2;
    const float* q1 = p1 + (size_t)n * PN2;
    const float* q2 = p2 + (size_t)n * PN2;

    float sg = 0.f, s0 = 0.f, s1 = 0.f, s2 = 0.f;
    for (int i = lane; i < PN2; i += 32) {
        float a0 = q0[i]; Ah[i]           = (_Float16)a0; s0 += a0 * a0;
        float a1 = q1[i]; Ah[PN2 + i]     = (_Float16)a1; s1 += a1 * a1;
        float a2 = q2[i]; Ah[2 * PN2 + i] = (_Float16)a2; s2 += a2 * a2;
    }
    for (int i = lane; i < 128; i += 32) {         // pack gt pairs + duplicate
        float g0 = g[2 * i], g1 = g[2 * i + 1];
        sg += g0 * g0 + g1 * g1;
        v2h hp; hp[0] = (_Float16)g0; hp[1] = (_Float16)g1;
        unsigned int w = __builtin_bit_cast(unsigned int, hp);
        Bw[i] = w; Bw[i + 128] = w;
    }

    if (n + 1 < NPOLY)  // CDNA5 global_prefetch_b8 for the next polygon's gt
        __builtin_prefetch(gt + (size_t)(n + 1) * PN2, 0, 1);

    for (int off = 16; off > 0; off >>= 1) {       // exact f32 sums, wave32
        sg += __shfl_xor(sg, off, 32);
        s0 += __shfl_xor(s0, off, 32);
        s1 += __shfl_xor(s1, off, 32);
        s2 += __shfl_xor(s2, off, 32);
    }
    __syncthreads();

    const int m = lane & 15;    // A-row / D-column index
    const int h = lane >> 4;    // lane half-group

    v8f acc[8] = {};            // 8 shift-tiles of 16 columns each
    for (int kk = 0; kk < 8; ++kk) {               // K = 256 in steps of 32
        v16h a;                 // A layout: halves [base,base+16)
        const int abase = m * PN2 + kk * 32 + 16 * h;
        #pragma unroll
        for (int e = 0; e < 8; ++e)  a[e] = Ah[abase + e];
        #pragma unroll
        for (int e = 8; e < 16; ++e) a[e] = Ah[abase + e];

        #pragma unroll
        for (int st = 0; st < 8; ++st) {
            // fragment halves start at 32kk+16h+32st+2m (even) -> dword index:
            const int ws0 = 16 * kk + 8 * h + 16 * st + m;   // <= 247, +7 < 256
            union { v16h h16; unsigned int u[8]; } bu;
            #pragma unroll
            for (int i = 0; i < 8; ++i) bu.u[i] = Bw[ws0 + i];
            acc[st] = __builtin_amdgcn_wmma_f32_16x16x32_f16(
                          false, a, false, bu.h16, (short)0, acc[st], false, false);
        }
    }

    const float base0 = 0.5f * (s0 + sg);
    const float base1 = 0.5f * (s1 + sg);
    const float base2 = 0.5f * (s2 + sg);
    float m0 = 3.4e38f, m1 = 3.4e38f, m2 = 3.4e38f;
    #pragma unroll
    for (int st = 0; st < 8; ++st) {
        // lanes 0..15 (h==0) hold D rows 0..7 in acc[st][0..7]; rows 0..2 = preds
        const float d0 = (base0 - acc[st][0]) * (1.f / 128.f);
        const float d1 = (base1 - acc[st][1]) * (1.f / 128.f);
        const float d2 = (base2 - acc[st][2]) * (1.f / 128.f);
        m0 = fminf(m0, d0); m1 = fminf(m1, d1); m2 = fminf(m2, d2);
    }
    // min across the 16 shift-columns held by lanes 0..15 (xor<16 stays in group)
    for (int off = 8; off > 0; off >>= 1) {
        m0 = fminf(m0, __shfl_xor(m0, off, 32));
        m1 = fminf(m1, __shfl_xor(m1, off, 32));
        m2 = fminf(m2, __shfl_xor(m2, off, 32));
    }
    if (lane == 0) {
        minD[n]             = m0;
        minD[NPOLY + n]     = m1;
        minD[2 * NPOLY + n] = m2;
    }
}

// ---------------------------------------------------------------------------
// Kernel 3: deterministic finalize -> scalar loss.
// ---------------------------------------------------------------------------
__global__ __launch_bounds__(256) void finalize_kernel(
    const float* __restrict__ partial, const float* __restrict__ minD,
    float* __restrict__ out)
{
    __shared__ float s[40];
    const int t = threadIdx.x;
    if (t < 4) {                               // global sums: cls,norm,ang,msum
        float a = 0.f;
        for (int b2 = 0; b2 < 1024; ++b2) a += partial[b2 * 8 + t];
        s[t] = a;
    } else if (t < 28) {                       // per-image npos/psum/nsum
        const int img = (t - 4) / 3, q = 4 + (t - 4) % 3;
        float a = 0.f;
        for (int b2 = img * 128; b2 < (img + 1) * 128; ++b2)
            a += partial[b2 * 8 + q];
        s[t] = a;
    } else if (t < 31) {                       // per-pred-set poly sums
        const int pt = t - 28;
        float a = 0.f;
        for (int n2 = 0; n2 < NPOLY; ++n2) a += minD[pt * NPOLY + n2];
        s[t] = a;
    }
    __syncthreads();
    if (t == 0) {
        const float cls = s[0] / ((float)NIMG * (float)HW);
        const float nrm = s[1] / ((float)NIMG * 512.f);
        const float ang = s[2] / fmaxf(s[3], 1.f);
        float dis = 0.f;
        for (int i = 0; i < NIMG; ++i) {
            const float np = s[4 + i * 3 + 0];
            const float ps = s[4 + i * 3 + 1];
            const float ns = s[4 + i * 3 + 2];
            const float k  = fminf((float)HW - np, 3.f * np);
            dis += ps / fmaxf(np, 1.f) + ns / fmaxf(k, 1.f);
        }
        dis *= (1.f / (float)NIMG);
        const float pl = (s[28] + s[29] + s[30]) * (1.f / (3.f * (float)NPOLY));
        out[0] = cls + dis + nrm + ang + pl;
    }
}

extern "C" void kernel_launch(void* const* d_in, const int* in_sizes, int n_in,
                              void* d_out, int out_size, void* d_ws, size_t ws_size,
                              hipStream_t stream) {
    const float* fy     = (const float*)d_in[0];
    const float* py0    = (const float*)d_in[1];
    const float* py1    = (const float*)d_in[2];
    const float* py2    = (const float*)d_in[3];
    const float* gtp    = (const float*)d_in[4];
    const int*   trainm = (const int*)d_in[5];
    const int*   trm    = (const int*)d_in[6];
    const float* dfld   = (const float*)d_in[7];
    const float* dirf   = (const float*)d_in[8];
    const float* wmat   = (const float*)d_in[9];

    float* ws      = (float*)d_ws;
    float* partial = ws;            // 1024 * 8 floats
    float* minD    = ws + 8192;     // 3 * 1024 floats

    field_losses_kernel<<<1024, 256, 0, stream>>>(fy, trainm, trm, dfld, dirf,
                                                  wmat, partial);
    poly_match_wmma_kernel<<<NPOLY, 32, 0, stream>>>(py0, py1, py2, gtp, minD);
    finalize_kernel<<<1, 256, 0, stream>>>(partial, minD, (float*)d_out);
}
